// PatchedVisionExpertAttention_90975997264557
// MI455X (gfx1250) — matrix operand order repaired and checked by reading, not compiled, not run on hardware
//
#include <hip/hip_runtime.h>
#include <math.h>

// ---------------- problem constants ----------------
#define T_TOK   2048
#define HID     4096
#define NHEAD   32
#define NKV     8
#define HDIM    128
#define QKVC    6144            // HID + 2*KV*D
#define KOFF    4096
#define VOFF    5120
#define BLOCK   64
#define NSEQ    4
#define CACHE_ELEMS (64*64*8*128)

typedef __attribute__((ext_vector_type(16))) __bf16 v16bf;
typedef __attribute__((ext_vector_type(8)))  float  v8f;
typedef __attribute__((ext_vector_type(2)))  float  v2f;
typedef __attribute__((ext_vector_type(2)))  __bf16 v2bf;

union BfFrag { v16bf v; unsigned int u[8]; };

// float pair -> packed bf16 pair via native convert (v_cvt_pk_bf16_f32)
__device__ __forceinline__ unsigned int pack2(float lo, float hi) {
  v2f x = {lo, hi};
  v2bf b = __builtin_convertvector(x, v2bf);
  return __builtin_bit_cast(unsigned int, b);
}
__device__ __forceinline__ unsigned short f2bf1(float f) {
  __bf16 b = (__bf16)f;
  return __builtin_bit_cast(unsigned short, b);
}
// 8 consecutive f32 -> 4 packed bf16 pairs
__device__ __forceinline__ uint4 pack8(float4 a, float4 b) {
  uint4 r;
  r.x = pack2(a.x, a.y); r.y = pack2(a.z, a.w);
  r.z = pack2(b.x, b.y); r.w = pack2(b.z, b.w);
  return r;
}

// A fragment (16x32 bf16) from a 16B-aligned row of 16 packed pairs.
__device__ __forceinline__ v16bf a_frag_lds(const uint4* row, int half) {
  BfFrag f;
  uint4 q0 = row[half];
  uint4 q1 = row[2 + half];
  f.u[0] = q0.x; f.u[1] = q0.y; f.u[2] = q0.z; f.u[3] = q0.w;
  f.u[4] = q1.x; f.u[5] = q1.y; f.u[6] = q1.z; f.u[7] = q1.w;
  return f.v;
}
// B fragment (32x16 bf16): u[v] = pair(4*half + v) for v<4, pair(8 + 4*half + v-4) for v>=4, from a 16B-aligned row.
__device__ __forceinline__ v16bf b_frag_lds(const uint4* row, int half) {
  BfFrag f;
  uint4 q0 = row[half];
  uint4 q1 = row[2 + half];
  f.u[0] = q0.x; f.u[1] = q0.y; f.u[2] = q0.z; f.u[3] = q0.w;
  f.u[4] = q1.x; f.u[5] = q1.y; f.u[6] = q1.z; f.u[7] = q1.w;
  return f.v;
}
// A fragment from contiguous f32 in global (16B-aligned); kb = half?8:0
__device__ __forceinline__ v16bf a_frag_f32v(const float* row, int kb) {
  const float4* p0 = (const float4*)(row + kb);
  const float4* p1 = (const float4*)(row + 16 + kb);
  BfFrag f;
  uint4 a = pack8(p0[0], p0[1]);
  uint4 b = pack8(p1[0], p1[1]);
  f.u[0] = a.x; f.u[1] = a.y; f.u[2] = a.z; f.u[3] = a.w;
  f.u[4] = b.x; f.u[5] = b.y; f.u[6] = b.z; f.u[7] = b.w;
  return f.v;
}

// ---------------- stable expert partition: vision tokens first --------------
__global__ void partition_kernel(const unsigned char* __restrict__ vmask,
                                 int* __restrict__ perm) {
  __shared__ int sdata[256];
  const int tid = threadIdx.x;

  int local = 0;
  for (int c = 0; c < 8; ++c) local += vmask[c * 256 + tid] ? 1 : 0;
  sdata[tid] = local;
  __syncthreads();
  for (int s = 128; s > 0; s >>= 1) {
    if (tid < s) sdata[tid] += sdata[tid + s];
    __syncthreads();
  }
  const int nv = sdata[0];
  __syncthreads();

  int voff = 0, loff = nv;
  for (int c = 0; c < 8; ++c) {
    int t = c * 256 + tid;
    int isv = vmask[t] ? 1 : 0;
    sdata[tid] = isv;
    __syncthreads();
    for (int s = 1; s < 256; s <<= 1) {
      int val = (tid >= s) ? sdata[tid - s] : 0;
      __syncthreads();
      sdata[tid] += val;
      __syncthreads();
    }
    int incl = sdata[tid];
    int excl = incl - isv;
    int chunkV = sdata[255];
    int pos = isv ? (voff + excl) : (loff + tid - excl);
    perm[pos] = t;
    voff += chunkV;
    loff += 256 - chunkV;
    __syncthreads();
  }
}

// ---------------- expert-routed GEMM on permuted rows ------------------------
// out[perm[r], :] = A[perm[r], :] @ (vmask[perm[r]] ? Wv : Wl)
// block = (32,8) = 8 waves, 64x64 tile of permuted rows; grid = (N/64, T/64).
// Rows are expert-sorted: blocks are expert-uniform except <=1 boundary block.
// The uniform/dual decision is hoisted OUT of the K-loop (block-uniform).
__global__ void expert_gemm_kernel(const float* __restrict__ A,
                                   const float* __restrict__ Wv,
                                   const float* __restrict__ Wl,
                                   const unsigned char* __restrict__ vmask,
                                   const int* __restrict__ perm,
                                   float* __restrict__ out,
                                   int K, int N) {
  const int lane = threadIdx.x;
  const int w    = threadIdx.y;                  // wave 0..7
  const int tid  = w * 32 + lane;
  const int row0 = blockIdx.y * 64;
  const int col0 = blockIdx.x * 64;
  const int half = lane >> 4;
  const int l15  = lane & 15;
  const int mr   = (w & 3) * 16;                 // wave's 16-row subtile
  const int nc   = (w >> 2) * 32;                // wave's 32-col subtile

  __shared__ __align__(16) uint4        Alds[64][5];   // packed-pair rows (80B stride)
  __shared__ __align__(16) unsigned int B0u[64][20];   // [n][pair]
  __shared__ __align__(16) unsigned int B1u[64][20];
  __shared__ int dualFlag;

  if (tid == 0) dualFlag = 0;
  __syncthreads();
  const unsigned char m0 = vmask[perm[row0]];
  if (vmask[perm[row0 + (tid & 63)]] != m0) dualFlag = 1;
  __syncthreads();
  const bool dual = dualFlag != 0;

  v8f acc0[2] = {};
  v8f acc1[2] = {};

  const int arow = tid >> 2;                     // 0..63
  const int aqi  = tid & 3;                      // 8-float chunk of K-slab
  const float* aptr = A + (size_t)perm[row0 + arow] * K + aqi * 8;

  const int bp  = tid >> 4;                      // pair 0..15
  const int bn0 = (tid & 15) * 4;                // column group of 4

  if (!dual) {
    // ---------- uniform block: single expert, branch-free K-loop ----------
    const float* W0 = m0 ? Wv : Wl;
    for (int kc = 0; kc < K; kc += 32) {
      {
        const float4* p = (const float4*)(aptr + kc);
        Alds[arow][aqi] = pack8(p[0], p[1]);
      }
      {
        size_t g0 = (size_t)(kc + 2 * bp) * N + col0 + bn0;
        float4 w0 = *(const float4*)(W0 + g0);
        float4 w1 = *(const float4*)(W0 + g0 + N);
        B0u[bn0 + 0][bp] = pack2(w0.x, w1.x);
        B0u[bn0 + 1][bp] = pack2(w0.y, w1.y);
        B0u[bn0 + 2][bp] = pack2(w0.z, w1.z);
        B0u[bn0 + 3][bp] = pack2(w0.w, w1.w);
        if (kc + 32 < K) {
          __builtin_prefetch(aptr + kc + 32, 0, 0);
          __builtin_prefetch(W0 + g0 + (size_t)32 * N, 0, 0);
        }
      }
      __syncthreads();
      v16bf a = a_frag_lds(&Alds[mr + l15][0], half);
#pragma unroll
      for (int ct = 0; ct < 2; ++ct) {
        int n = nc + ct * 16 + l15;
        v16bf b0 = b_frag_lds((const uint4*)&B0u[n][0], half);
        acc0[ct] = __builtin_amdgcn_wmma_f32_16x16x32_bf16(false, a, false, b0, (short)0, acc0[ct], false, false);
      }
      __syncthreads();
    }
  } else {
    // ---------- boundary block: both experts, branch-free K-loop ----------
    for (int kc = 0; kc < K; kc += 32) {
      {
        const float4* p = (const float4*)(aptr + kc);
        Alds[arow][aqi] = pack8(p[0], p[1]);
      }
      {
        size_t g0 = (size_t)(kc + 2 * bp) * N + col0 + bn0;
        float4 w0 = *(const float4*)(Wv + g0);
        float4 w1 = *(const float4*)(Wv + g0 + N);
        B0u[bn0 + 0][bp] = pack2(w0.x, w1.x);
        B0u[bn0 + 1][bp] = pack2(w0.y, w1.y);
        B0u[bn0 + 2][bp] = pack2(w0.z, w1.z);
        B0u[bn0 + 3][bp] = pack2(w0.w, w1.w);
        float4 u0 = *(const float4*)(Wl + g0);
        float4 u1 = *(const float4*)(Wl + g0 + N);
        B1u[bn0 + 0][bp] = pack2(u0.x, u1.x);
        B1u[bn0 + 1][bp] = pack2(u0.y, u1.y);
        B1u[bn0 + 2][bp] = pack2(u0.z, u1.z);
        B1u[bn0 + 3][bp] = pack2(u0.w, u1.w);
      }
      __syncthreads();
      v16bf a = a_frag_lds(&Alds[mr + l15][0], half);
#pragma unroll
      for (int ct = 0; ct < 2; ++ct) {
        int n = nc + ct * 16 + l15;
        v16bf b0 = b_frag_lds((const uint4*)&B0u[n][0], half);
        v16bf b1 = b_frag_lds((const uint4*)&B1u[n][0], half);
        acc0[ct] = __builtin_amdgcn_wmma_f32_16x16x32_bf16(false, a, false, b0, (short)0, acc0[ct], false, false);
        acc1[ct] = __builtin_amdgcn_wmma_f32_16x16x32_bf16(false, a, false, b1, (short)0, acc1[ct], false, false);
      }
      __syncthreads();
    }
  }

  // ---- epilogue: scatter to natural token order ----
  int prow[8];
  unsigned char msel[8];
#pragma unroll
  for (int r = 0; r < 8; ++r) {
    prow[r] = perm[row0 + mr + r + 8 * half];
    msel[r] = vmask[prow[r]];
  }
#pragma unroll
  for (int ct = 0; ct < 2; ++ct) {
#pragma unroll
    for (int r = 0; r < 8; ++r) {
      float val = dual ? (msel[r] ? acc0[ct][r] : acc1[ct][r]) : acc0[ct][r];
      out[(size_t)prow[r] * N + col0 + nc + ct * 16 + l15] = val;
    }
  }
}

// ---------------- copy input caches into output caches ----------------
__global__ void cache_copy_kernel(const float* __restrict__ kc_in,
                                  const float* __restrict__ vc_in,
                                  float* __restrict__ kc_out,
                                  float* __restrict__ vc_out, int n) {
  int i = blockIdx.x * blockDim.x + threadIdx.x;
  int stride = gridDim.x * blockDim.x;
  for (; i < n; i += stride) {
    kc_out[i] = kc_in[i];
    vc_out[i] = vc_in[i];
  }
}

// ---------------- RoPE (in-place on ws QKV) + paged KV scatter -------------
__global__ void rope_scatter_kernel(float* __restrict__ qkv,
                                    const int* __restrict__ pos_ids,
                                    const int* __restrict__ seq_ids,
                                    const int* __restrict__ block_offsets, int maxb,
                                    const float* __restrict__ inv_freq,
                                    float* __restrict__ kc_out,
                                    float* __restrict__ vc_out) {
  const int t = blockIdx.x;
  const int tid = threadIdx.x;
  const int pos = pos_ids[t];
  const int seq = seq_ids[t];
  const int blk = block_offsets[seq * maxb + pos / BLOCK];
  const int slot = pos % BLOCK;
  float* base = qkv + (size_t)t * QKVC;

  for (int idx = tid; idx < NHEAD * 64; idx += blockDim.x) {
    int h = idx >> 6, i = idx & 63;
    float f = (float)pos * inv_freq[i];
    float c = cosf(f), s = sinf(f);
    float* p = base + h * HDIM;
    float x0 = p[i], x1 = p[i + 64];
    p[i]      = x0 * c - x1 * s;
    p[i + 64] = x1 * c + x0 * s;
  }
  for (int idx = tid; idx < NKV * 64; idx += blockDim.x) {
    int h = idx >> 6, i = idx & 63;
    float f = (float)pos * inv_freq[i];
    float c = cosf(f), s = sinf(f);
    float* p = base + KOFF + h * HDIM;
    float x0 = p[i], x1 = p[i + 64];
    p[i]      = x0 * c - x1 * s;
    p[i + 64] = x1 * c + x0 * s;
  }
  __syncthreads();
  size_t crow = ((size_t)blk * BLOCK + slot) * (NKV * HDIM);
  for (int idx = tid; idx < NKV * HDIM; idx += blockDim.x) {
    kc_out[crow + idx] = base[KOFF + idx];
    vc_out[crow + idx] = base[VOFF + idx];
  }
}

// ---------------- flash GQA attention ---------------------------------------
// block = (32,4): one KV-head group; wave w = query head kvh*4+w, 16 queries.
// K and V tiles staged once in LDS, shared by the 4 heads. grid = (T/16, NKV)
__global__ void attn_kernel(const float* __restrict__ qkv,
                            const int* __restrict__ seq_ids,
                            const int* __restrict__ pos_ids,
                            float* __restrict__ ctx) {
  const int lane = threadIdx.x;
  const int w    = threadIdx.y;
  const int tid  = w * 32 + lane;
  const int qt   = blockIdx.x;
  const int kvh  = blockIdx.y;
  const int h    = kvh * 4 + w;
  const int qbase = qt * 16;
  const int half = lane >> 4;
  const int l15  = lane & 15;
  const int kbA  = half ? 8 : 0;
  const float scale = 0.08838834764831845f;      // 1/sqrt(128)

  __shared__ __align__(16) unsigned short Klds[32][144];   // bf16 K rows, 288B stride
  __shared__ __align__(16) unsigned int   Vp[128][20];     // packed V pairs [d][pair]
  __shared__ __align__(16) unsigned short Plds[4][16][32]; // per-wave bf16 P tile

  const float* qrow = qkv + (size_t)(qbase + l15) * QKVC + h * HDIM;
  v16bf qa[4];
#pragma unroll
  for (int dk = 0; dk < 4; ++dk) qa[dk] = a_frag_f32v(qrow + dk * 32, kbA);

  int qseq[8], qpos[8];
#pragma unroll
  for (int r = 0; r < 8; ++r) {
    int q = qbase + r + 8 * half;
    qseq[r] = seq_ids[q];
    qpos[r] = pos_ids[q];
  }

  float mrow[8], lrow[8];
  v8f o[8] = {};
#pragma unroll
  for (int r = 0; r < 8; ++r) { mrow[r] = -INFINITY; lrow[r] = 0.0f; }

  const int nkt = (qbase + 16 + 31) >> 5;
  for (int kt = 0; kt < nkt; ++kt) {
    const int kbase = kt * 32;

#pragma unroll
    for (int it = 0; it < 4; ++it) {
      int idx = it * 128 + tid;                  // 512 = 32 keys x 16 d-groups of 8
      int key = idx >> 4;
      int dg  = idx & 15;
      const float4* src = (const float4*)(qkv + (size_t)(kbase + key) * QKVC + KOFF + kvh * HDIM);
      ((uint4*)&Klds[key][0])[dg] = pack8(src[2 * dg], src[2 * dg + 1]);
    }
#pragma unroll
    for (int it = 0; it < 4; ++it) {
      int idx = it * 128 + tid;                  // 512 = 16 pairs x 32 d-groups of 4
      int p  = idx >> 5;
      int dg = idx & 31;
      const float4* r0 = (const float4*)(qkv + (size_t)(kbase + 2 * p)     * QKVC + VOFF + kvh * HDIM) + dg;
      const float4* r1 = (const float4*)(qkv + (size_t)(kbase + 2 * p + 1) * QKVC + VOFF + kvh * HDIM) + dg;
      float4 a = *r0, b = *r1;
      int d0 = dg * 4;
      Vp[d0 + 0][p] = pack2(a.x, b.x);
      Vp[d0 + 1][p] = pack2(a.y, b.y);
      Vp[d0 + 2][p] = pack2(a.z, b.z);
      Vp[d0 + 3][p] = pack2(a.w, b.w);
    }
    __syncthreads();

    v8f s0 = {}, s1 = {};
#pragma unroll
    for (int dk = 0; dk < 4; ++dk) {
      v16bf b0 = b_frag_lds((const uint4*)&Klds[l15][0]      + dk * 4, half);
      v16bf b1 = b_frag_lds((const uint4*)&Klds[16 + l15][0] + dk * 4, half);
      s0 = __builtin_amdgcn_wmma_f32_16x16x32_bf16(false, qa[dk], false, b0, (short)0, s0, false, false);
      s1 = __builtin_amdgcn_wmma_f32_16x16x32_bf16(false, qa[dk], false, b1, (short)0, s1, false, false);
    }

    const int kseq0 = seq_ids[kbase + l15];
    const int kpos0 = pos_ids[kbase + l15];
    const int kseq1 = seq_ids[kbase + 16 + l15];
    const int kpos1 = pos_ids[kbase + 16 + l15];
#pragma unroll
    for (int r = 0; r < 8; ++r) {
      bool ok0 = (kseq0 == qseq[r]) && (kpos0 <= qpos[r]);
      bool ok1 = (kseq1 == qseq[r]) && (kpos1 <= qpos[r]);
      s0[r] = ok0 ? s0[r] * scale : -1.0e9f;
      s1[r] = ok1 ? s1[r] * scale : -1.0e9f;
    }

#pragma unroll
    for (int r = 0; r < 8; ++r) {
      float mx = fmaxf(s0[r], s1[r]);
#pragma unroll
      for (int m = 1; m < 16; m <<= 1) mx = fmaxf(mx, __shfl_xor(mx, m, 32));
      float newm = fmaxf(mrow[r], mx);
      float alpha = __expf(mrow[r] - newm);
      float p0 = __expf(s0[r] - newm);
      float p1 = __expf(s1[r] - newm);
      float sum = p0 + p1;
#pragma unroll
      for (int m = 1; m < 16; m <<= 1) sum += __shfl_xor(sum, m, 32);
      lrow[r] = lrow[r] * alpha + sum;
      mrow[r] = newm;
#pragma unroll
      for (int dc = 0; dc < 8; ++dc) o[dc][r] *= alpha;
      int M = r + 8 * half;
      Plds[w][M][l15]      = f2bf1(p0);
      Plds[w][M][16 + l15] = f2bf1(p1);
    }

    {
      v16bf pa = a_frag_lds((const uint4*)&Plds[w][l15][0], half);
#pragma unroll
      for (int dc = 0; dc < 8; ++dc) {
        v16bf bv = b_frag_lds((const uint4*)&Vp[dc * 16 + l15][0], half);
        o[dc] = __builtin_amdgcn_wmma_f32_16x16x32_bf16(false, pa, false, bv, (short)0, o[dc], false, false);
      }
    }
    __syncthreads();
  }

#pragma unroll
  for (int dc = 0; dc < 8; ++dc) {
#pragma unroll
    for (int r = 0; r < 8; ++r) {
      int q = qbase + r + 8 * half;
      ctx[(size_t)q * HID + h * HDIM + dc * 16 + l15] = o[dc][r] / lrow[r];
    }
  }
}

// ---------------- launch ----------------
extern "C" void kernel_launch(void* const* d_in, const int* in_sizes, int n_in,
                              void* d_out, int out_size, void* d_ws, size_t ws_size,
                              hipStream_t stream) {
  const float* X     = (const float*)d_in[0];
  const int*   pos   = (const int*)d_in[1];
  const int*   seq   = (const int*)d_in[2];
  const unsigned char* vmask = (const unsigned char*)d_in[3];
  const int*   bofs  = (const int*)d_in[4];
  const float* kc_in = (const float*)d_in[5];
  const float* vc_in = (const float*)d_in[6];
  const float* Wqv   = (const float*)d_in[7];
  const float* Wql   = (const float*)d_in[8];
  const float* Wdv   = (const float*)d_in[9];
  const float* Wdl   = (const float*)d_in[10];
  const float* invf  = (const float*)d_in[11];
  const int maxb = in_sizes[4] / NSEQ;

  float* out_attn = (float*)d_out;
  float* out_kc   = out_attn + (size_t)T_TOK * HID;
  float* out_vc   = out_kc + CACHE_ELEMS;

  float* qkv  = (float*)d_ws;                       // [T,QKVC] f32
  float* ctx  = qkv + (size_t)T_TOK * QKVC;         // [T,HID]  f32
  int*   perm = (int*)(ctx + (size_t)T_TOK * HID);  // [T] expert-sorted order

  partition_kernel<<<1, 256, 0, stream>>>(vmask, perm);

  expert_gemm_kernel<<<dim3(QKVC / 64, T_TOK / 64), dim3(32, 8), 0, stream>>>(
      X, Wqv, Wql, vmask, perm, qkv, HID, QKVC);

  cache_copy_kernel<<<4096, 256, 0, stream>>>(kc_in, vc_in, out_kc, out_vc, CACHE_ELEMS);

  rope_scatter_kernel<<<T_TOK, 256, 0, stream>>>(qkv, pos, seq, bofs, maxb, invf,
                                                 out_kc, out_vc);

  attn_kernel<<<dim3(T_TOK / 16, NKV), dim3(32, 4), 0, stream>>>(qkv, seq, pos, ctx);

  expert_gemm_kernel<<<dim3(HID / 64, T_TOK / 64), dim3(32, 8), 0, stream>>>(
      ctx, Wdv, Wdl, vmask, perm, out_attn, HID, HID);
}
